// PointCloudVQVAE_63806034150161
// MI455X (gfx1250) — matrix operand order repaired and checked by reading, not compiled, not run on hardware
//
#include <hip/hip_runtime.h>
#include <hip/hip_bf16.h>
#include <math.h>

#define LN_EPS 1e-5f
#define VQ_BETA 0.25f

typedef __attribute__((ext_vector_type(16))) __bf16 v16bf;
typedef __attribute__((ext_vector_type(8)))  __bf16 v8bf;
typedef __attribute__((ext_vector_type(8)))  float  v8f;

static __device__ __forceinline__ unsigned short f2bf(float f) {
  unsigned int u = __float_as_uint(f);
  unsigned int r = (u + 0x7FFFu + ((u >> 16) & 1u)) >> 16; // RNE
  return (unsigned short)r;
}
static __device__ __forceinline__ float gelu_exact(float x) {
  return 0.5f * x * (1.0f + erff(x * 0.70710678118654752f));
}

// A fragment (16x32 bf16, MxK): lane = half*16 + m ; element e -> K=ks*32+16*half+e.
// Activations stored row-major [m][k] in LDS.
static __device__ __forceinline__ v16bf load_afrag(const unsigned short* act, int m,
                                                   int Kstride, int ks, int half) {
  const unsigned short* p = act + m * Kstride + ks * 32 + half * 16;
  v8bf lo = *(const v8bf*)p;
  v8bf hi = *(const v8bf*)(p + 8);
  return __builtin_shufflevector(lo, hi, 0,1,2,3,4,5,6,7,8,9,10,11,12,13,14,15);
}
// B fragment (32x16 bf16, KxN): lane = half*16 + n ; element e -> K=ks*32+16*half+e.
// Weights stored transposed [n][k] in LDS so this is one contiguous 32B load.
static __device__ __forceinline__ v16bf load_bfrag(const unsigned short* wt, int n,
                                                   int Kstride, int ks, int half) {
  return *(const v16bf*)(wt + n * Kstride + ks * 32 + half * 16);
}

// ---------------------------------------------------------------------------
// Encoder: fused 3->64->128->256 MLP (LN+GELU each layer) + per-block max pool
// grid = 512 blocks x 256 threads (8 waves); 4 tiles of 128 points per block.
// ---------------------------------------------------------------------------
__launch_bounds__(256, 1)
__global__ void vqvae_encoder_kernel(
    const float* __restrict__ pts,
    const float* __restrict__ W1, const float* __restrict__ B1,
    const float* __restrict__ G1w, const float* __restrict__ G1b,
    const float* __restrict__ W2, const float* __restrict__ B2,
    const float* __restrict__ G2w, const float* __restrict__ G2b,
    const float* __restrict__ W3, const float* __restrict__ B3,
    const float* __restrict__ G3w, const float* __restrict__ G3b,
    float* __restrict__ partials)
{
  __shared__ __align__(32) float sW1[192];
  __shared__ float sB1[64], sG1w[64], sG1b[64];
  __shared__ __align__(32) unsigned short sW2t[128 * 64];    // [n][k] bf16
  __shared__ float sB2[128], sG2w[128], sG2b[128];
  __shared__ __align__(32) unsigned short sW3t[256 * 128];   // [n][k] bf16
  __shared__ float sB3[256], sG3w[256], sG3b[256];
  __shared__ __align__(32) unsigned short sAct1[8 * 16 * 64];  // per-wave [m][k]
  __shared__ __align__(32) unsigned short sAct2[8 * 16 * 128]; // per-wave [m][k]
  __shared__ float sPool[8 * 256];

  const int tid = threadIdx.x;

  // Stage weights (convert to bf16, transpose W2/W3 for contiguous B-frags)
  for (int i = tid; i < 192; i += 256) sW1[i] = W1[i];
  for (int i = tid; i < 64; i += 256) { sB1[i] = B1[i]; sG1w[i] = G1w[i]; sG1b[i] = G1b[i]; }
  for (int i = tid; i < 128; i += 256) { sB2[i] = B2[i]; sG2w[i] = G2w[i]; sG2b[i] = G2b[i]; }
  for (int i = tid; i < 256; i += 256) { sB3[i] = B3[i]; sG3w[i] = G3w[i]; sG3b[i] = G3b[i]; }
  for (int i = tid; i < 64 * 128; i += 256) { int k = i >> 7, n = i & 127; sW2t[n * 64 + k] = f2bf(W2[i]); }
  for (int i = tid; i < 128 * 256; i += 256) { int k = i >> 8, n = i & 255; sW3t[n * 128 + k] = f2bf(W3[i]); }
  __syncthreads();

  const int w = tid >> 5;
  const int lane = tid & 31;
  const int col = lane & 15;
  const int half = lane >> 4;
  unsigned short* act1 = sAct1 + w * 1024;
  unsigned short* act2 = sAct2 + w * 2048;

  float rmax[16];
  #pragma unroll
  for (int nt = 0; nt < 16; nt++) rmax[nt] = -3.402823466e38f;

  for (int it = 0; it < 4; it++) {
    const int base = blockIdx.x * 512 + it * 128 + w * 16;

    // ---- layer 1: 3 -> 64, one point per lane (lanes 0..15), LN+GELU in-lane ----
    if (lane < 16) {
      const int p = base + lane;
      const float x0 = pts[p * 3 + 0], x1 = pts[p * 3 + 1], x2 = pts[p * 3 + 2];
      float h[64]; float s = 0.f, s2 = 0.f;
      #pragma unroll
      for (int j = 0; j < 64; j++) {
        float v = fmaf(x0, sW1[j], fmaf(x1, sW1[64 + j], fmaf(x2, sW1[128 + j], sB1[j])));
        h[j] = v; s += v; s2 += v * v;
      }
      const float mu = s * (1.f / 64.f);
      const float var = s2 * (1.f / 64.f) - mu * mu;
      const float inv = rsqrtf(var + LN_EPS);
      #pragma unroll
      for (int j = 0; j < 64; j++) {
        float y = (h[j] - mu) * inv * sG1w[j] + sG1b[j];
        act1[lane * 64 + j] = f2bf(gelu_exact(y));
      }
    }

    // ---- layer 2: 64 -> 128 via v_wmma_f32_16x16x32_bf16 ----
    v8f acc2[8];
    #pragma unroll
    for (int nt = 0; nt < 8; nt++) { v8f z = {0.f,0.f,0.f,0.f,0.f,0.f,0.f,0.f}; acc2[nt] = z; }
    #pragma unroll
    for (int ks = 0; ks < 2; ks++) {
      v16bf a = load_afrag(act1, col, 64, ks, half);
      #pragma unroll
      for (int nt = 0; nt < 8; nt++) {
        v16bf b = load_bfrag(sW2t, nt * 16 + col, 64, ks, half);
        acc2[nt] = __builtin_amdgcn_wmma_f32_16x16x32_bf16(false, a, false, b,
                                                           (short)0, acc2[nt], false, false);
      }
    }
    { // bias + LN (rows v+8*half, 128 cols) + GELU -> act2 bf16
      float rs[8], rs2[8];
      #pragma unroll
      for (int v = 0; v < 8; v++) { rs[v] = 0.f; rs2[v] = 0.f; }
      #pragma unroll
      for (int nt = 0; nt < 8; nt++)
        #pragma unroll
        for (int v = 0; v < 8; v++) {
          float x = acc2[nt][v] + sB2[nt * 16 + col];
          rs[v] += x; rs2[v] += x * x;
        }
      #pragma unroll
      for (int v = 0; v < 8; v++)
        #pragma unroll
        for (int m = 1; m < 16; m <<= 1) {
          rs[v]  += __shfl_xor(rs[v],  m);
          rs2[v] += __shfl_xor(rs2[v], m);
        }
      #pragma unroll
      for (int v = 0; v < 8; v++) {
        const float mu  = rs[v]  * (1.f / 128.f);
        const float var = rs2[v] * (1.f / 128.f) - mu * mu;
        const float inv = rsqrtf(var + LN_EPS);
        #pragma unroll
        for (int nt = 0; nt < 8; nt++) {
          const int n = nt * 16 + col;
          float x = acc2[nt][v] + sB2[n];
          float y = (x - mu) * inv * sG2w[n] + sG2b[n];
          act2[(v + 8 * half) * 128 + n] = f2bf(gelu_exact(y));
        }
      }
    }

    // ---- layer 3: 128 -> 256 via v_wmma_f32_16x16x32_bf16 ----
    v8f acc3[16];
    #pragma unroll
    for (int nt = 0; nt < 16; nt++) { v8f z = {0.f,0.f,0.f,0.f,0.f,0.f,0.f,0.f}; acc3[nt] = z; }
    #pragma unroll
    for (int ks = 0; ks < 4; ks++) {
      v16bf a = load_afrag(act2, col, 128, ks, half);
      #pragma unroll
      for (int nt = 0; nt < 16; nt++) {
        v16bf b = load_bfrag(sW3t, nt * 16 + col, 128, ks, half);
        acc3[nt] = __builtin_amdgcn_wmma_f32_16x16x32_bf16(false, a, false, b,
                                                           (short)0, acc3[nt], false, false);
      }
    }
    { // bias + LN over 256 cols + GELU + running max-pool
      float rs[8], rs2[8];
      #pragma unroll
      for (int v = 0; v < 8; v++) { rs[v] = 0.f; rs2[v] = 0.f; }
      #pragma unroll
      for (int nt = 0; nt < 16; nt++)
        #pragma unroll
        for (int v = 0; v < 8; v++) {
          float x = acc3[nt][v] + sB3[nt * 16 + col];
          rs[v] += x; rs2[v] += x * x;
        }
      #pragma unroll
      for (int v = 0; v < 8; v++)
        #pragma unroll
        for (int m = 1; m < 16; m <<= 1) {
          rs[v]  += __shfl_xor(rs[v],  m);
          rs2[v] += __shfl_xor(rs2[v], m);
        }
      float mloc[16];
      #pragma unroll
      for (int nt = 0; nt < 16; nt++) mloc[nt] = -3.402823466e38f;
      #pragma unroll
      for (int v = 0; v < 8; v++) {
        const float mu  = rs[v]  * (1.f / 256.f);
        const float var = rs2[v] * (1.f / 256.f) - mu * mu;
        const float inv = rsqrtf(var + LN_EPS);
        #pragma unroll
        for (int nt = 0; nt < 16; nt++) {
          const int n = nt * 16 + col;
          float x = acc3[nt][v] + sB3[n];
          float g = gelu_exact((x - mu) * inv * sG3w[n] + sG3b[n]);
          mloc[nt] = fmaxf(mloc[nt], g);
        }
      }
      #pragma unroll
      for (int nt = 0; nt < 16; nt++) {
        float mm = fmaxf(mloc[nt], __shfl_xor(mloc[nt], 16)); // merge halves (rows 0-7 / 8-15)
        rmax[nt] = fmaxf(rmax[nt], mm);
      }
    }
  }

  if (half == 0) {
    #pragma unroll
    for (int nt = 0; nt < 16; nt++) sPool[w * 256 + nt * 16 + col] = rmax[nt];
  }
  __syncthreads();
  if (tid < 256) {
    float m = -3.402823466e38f;
    #pragma unroll
    for (int ww = 0; ww < 8; ww++) m = fmaxf(m, sPool[ww * 256 + tid]);
    partials[blockIdx.x * 256 + tid] = m;
  }
}

// ---------------------------------------------------------------------------
// Tail: pool-reduce, z_e GEMV, 8-level residual VQ, decoder MLP, recon output.
// Single block of 1024 threads (work here is ~5 MFLOP, negligible).
// ---------------------------------------------------------------------------
__launch_bounds__(1024, 1)
__global__ void vqvae_tail_kernel(
    const float* __restrict__ partials, int nparts,
    const float* __restrict__ eW, const float* __restrict__ eB,
    const float* __restrict__ cbs,
    const float* __restrict__ dW0, const float* __restrict__ dW1, const float* __restrict__ dW2,
    const float* __restrict__ dB0, const float* __restrict__ dB1, const float* __restrict__ dB2,
    const float* __restrict__ dG0w, const float* __restrict__ dG1w, const float* __restrict__ dG2w,
    const float* __restrict__ dG0b, const float* __restrict__ dG1b, const float* __restrict__ dG2b,
    const float* __restrict__ oW, const float* __restrict__ oB,
    float* __restrict__ out)
{
  __shared__ float sPooled[256];
  __shared__ float sRes[256], sZq[256];
  __shared__ float sHa[512], sHb[512];
  __shared__ float sRed[1024];
  __shared__ float sSc[1024];
  __shared__ int   sIx[1024];
  const int tid = threadIdx.x;

  if (tid < 256) {
    float m = -3.402823466e38f;
    for (int b = 0; b < nparts; b++) m = fmaxf(m, partials[b * 256 + tid]);
    sPooled[tid] = m;
  }
  __syncthreads();

  if (tid < 256) { // z_e = pooled @ enc_out_w + enc_out_b
    float a = eB[tid];
    for (int d = 0; d < 256; d++) a = fmaf(sPooled[d], eW[d * 256 + tid], a);
    out[6144 + tid] = a;       // z_e
    sRes[tid] = a; sZq[tid] = 0.f;
  }
  __syncthreads();

  float vloss = 0.f;
  for (int l = 0; l < 8; l++) {
    const float* cb = cbs + l * 1024 * 256;
    const float* cj = cb + tid * 256;
    float dot = 0.f, c2 = 0.f;
    for (int d = 0; d < 256; d++) { float c = cj[d]; dot = fmaf(sRes[d], c, dot); c2 = fmaf(c, c, c2); }
    sSc[tid] = c2 - 2.f * dot;  // argmin unaffected by +||r||^2
    sIx[tid] = tid;
    __syncthreads();
    for (int o = 512; o > 0; o >>= 1) {
      if (tid < o) {
        float a = sSc[tid], b = sSc[tid + o];
        int ib = sIx[tid + o];
        if (b < a || (b == a && ib < sIx[tid])) { sSc[tid] = b; sIx[tid] = ib; }
      }
      __syncthreads();
    }
    const int idx = sIx[0];
    float contrib = 0.f;
    if (tid < 256) {
      float c = cb[idx * 256 + tid];
      float r = sRes[tid];
      float df = r - c;
      contrib = df * df;
      sZq[tid] += c;             // quantized_st == code numerically
      sRes[tid] = r - c;
    }
    sRed[tid] = contrib; __syncthreads();
    for (int o = 512; o > 0; o >>= 1) { if (tid < o) sRed[tid] += sRed[tid + o]; __syncthreads(); }
    vloss += (1.f + VQ_BETA) * (sRed[0] * (1.f / 256.f)); // cb-loss + BETA*commit (equal fwd)
    __syncthreads();
  }
  if (tid < 256) out[6400 + tid] = sZq[tid]; // z_q
  if (tid == 0)  out[6656] = vloss;          // vq_loss

  if (tid < 256) sHa[tid] = sZq[tid];
  __syncthreads();
  const float* Ws[3] = {dW0, dW1, dW2};
  const float* Bs[3] = {dB0, dB1, dB2};
  const float* Gw[3] = {dG0w, dG1w, dG2w};
  const float* Gb[3] = {dG0b, dG1b, dG2b};
  float* src = sHa; float* dst = sHb; int din = 256;
  for (int i = 0; i < 3; i++) {
    float acc = 0.f;
    if (tid < 512) {
      acc = Bs[i][tid];
      for (int d = 0; d < din; d++) acc = fmaf(src[d], Ws[i][d * 512 + tid], acc);
    }
    sRed[tid] = (tid < 512) ? acc : 0.f; __syncthreads();
    for (int o = 512; o > 0; o >>= 1) { if (tid < o) sRed[tid] += sRed[tid + o]; __syncthreads(); }
    const float mu = sRed[0] * (1.f / 512.f); __syncthreads();
    float dv = (tid < 512) ? (acc - mu) : 0.f;
    sRed[tid] = dv * dv; __syncthreads();
    for (int o = 512; o > 0; o >>= 1) { if (tid < o) sRed[tid] += sRed[tid + o]; __syncthreads(); }
    const float var = sRed[0] * (1.f / 512.f); __syncthreads();
    if (tid < 512) {
      float y = (acc - mu) * rsqrtf(var + LN_EPS) * Gw[i][tid] + Gb[i][tid];
      dst[tid] = gelu_exact(y);
    }
    __syncthreads();
    float* t = src; src = dst; dst = t; din = 512;
  }
  for (int o = tid; o < 6144; o += 1024) { // recon = h @ dec_out_w + dec_out_b
    float a = oB[o];
    for (int d = 0; d < 512; d++) a = fmaf(src[d], oW[d * 6144 + o], a);
    out[o] = a;
  }
}

extern "C" void kernel_launch(void* const* d_in, const int* in_sizes, int n_in,
                              void* d_out, int out_size, void* d_ws, size_t ws_size,
                              hipStream_t stream) {
  const float* pts   = (const float*)d_in[0];
  const float* eW1   = (const float*)d_in[1];
  const float* eW2   = (const float*)d_in[2];
  const float* eW3   = (const float*)d_in[3];
  const float* eB1   = (const float*)d_in[4];
  const float* eB2   = (const float*)d_in[5];
  const float* eB3   = (const float*)d_in[6];
  const float* eG1w  = (const float*)d_in[7];
  const float* eG2w  = (const float*)d_in[8];
  const float* eG3w  = (const float*)d_in[9];
  const float* eG1b  = (const float*)d_in[10];
  const float* eG2b  = (const float*)d_in[11];
  const float* eG3b  = (const float*)d_in[12];
  const float* eOutW = (const float*)d_in[13];
  const float* eOutB = (const float*)d_in[14];
  const float* cbs   = (const float*)d_in[15];
  const float* dW0   = (const float*)d_in[16];
  const float* dW1   = (const float*)d_in[17];
  const float* dW2   = (const float*)d_in[18];
  const float* dB0   = (const float*)d_in[19];
  const float* dB1   = (const float*)d_in[20];
  const float* dB2   = (const float*)d_in[21];
  const float* dG0w  = (const float*)d_in[22];
  const float* dG1w  = (const float*)d_in[23];
  const float* dG2w  = (const float*)d_in[24];
  const float* dG0b  = (const float*)d_in[25];
  const float* dG1b  = (const float*)d_in[26];
  const float* dG2b  = (const float*)d_in[27];
  const float* oW    = (const float*)d_in[28];
  const float* oB    = (const float*)d_in[29];
  float* out = (float*)d_out;
  float* partials = (float*)d_ws; // 512 blocks * 256 floats = 512 KB

  vqvae_encoder_kernel<<<512, 256, 0, stream>>>(
      pts, eW1, eB1, eG1w, eG1b, eW2, eB2, eG2w, eG2b, eW3, eB3, eG3w, eG3b, partials);
  vqvae_tail_kernel<<<1, 1024, 0, stream>>>(
      partials, 512, eOutW, eOutB, cbs, dW0, dW1, dW2, dB0, dB1, dB2,
      dG0w, dG1w, dG2w, dG0b, dG1b, dG2b, oW, oB, out);
}
